// SABlock_SPA_11742440587452
// MI455X (gfx1250) — matrix operand order repaired and checked by reading, not compiled, not run on hardware
//
#include <hip/hip_runtime.h>

// ---------------------------------------------------------------------------
// SPA attention block for MI455X (gfx1250), bf16 WMMA everywhere.
//   B=2, N=2048, C=768, H=12, D=64, SCALE=1/8, GAMMA=0.5, block period 512.
// Pipeline:
//   k0: fp32 -> bf16 conversion of x, w_qkv, w_out
//   k1: QKV GEMM, 32x64 register tile per wave (8 WMMA / k-step), epilogue
//       scatters Q[B,H,N,D], K[B,H,N,D], V^T[B,H,D,N] as bf16
//   k2: fused scores -> softmax -> mask -> softmax -> P@V per 16-query block;
//       fp32 scores (131KB) + bf16 probs (66KB) live in LDS, never touch HBM
//   k3: output projection GEMM (same 32x64 tiling) + bias, fp32 out
// ---------------------------------------------------------------------------

#define HD    64
#define NH    12
#define NC    768
#define NSEQ  2048
#define NB    2
#define NROWS (NB * NSEQ)   // 4096
#define F3C   (3 * NC)      // 2304
#define ATT_SCALE 0.125f
#define LDST  2052          // fp32 score row stride (2048+4) -> conflict-free
#define PBS   2056          // bf16 prob row stride (2048+8)  -> conflict-free

typedef __bf16 bf16_t;
typedef __attribute__((ext_vector_type(16))) __bf16 v16bf;
typedef __attribute__((ext_vector_type(8)))  __bf16 v8bf;
typedef __attribute__((ext_vector_type(8)))  float  v8f;

// B operand: lane holds 16 consecutive K values -> 2x16B contiguous
static __device__ __forceinline__ v16bf ld_cont16(const bf16_t* p) {
  v8bf lo = *reinterpret_cast<const v8bf*>(p);
  v8bf hi = *reinterpret_cast<const v8bf*>(p + 8);
  return __builtin_shufflevector(lo, hi, 0,1,2,3,4,5,6,7,8,9,10,11,12,13,14,15);
}
// A operand: lane holds 16 consecutive K values (8 bf16 at p, 8 bf16 at p+8)
static __device__ __forceinline__ v16bf ld_split8(const bf16_t* p) {
  v8bf lo = *reinterpret_cast<const v8bf*>(p);
  v8bf hi = *reinterpret_cast<const v8bf*>(p + 8);
  return __builtin_shufflevector(lo, hi, 0,1,2,3,4,5,6,7,8,9,10,11,12,13,14,15);
}
static __device__ __forceinline__ v8f wmma_bf16(v16bf a, v16bf b, v8f c) {
  return __builtin_amdgcn_wmma_f32_16x16x32_bf16(false, a, false, b,
                                                 (short)0, c, false, false);
}
static __device__ __forceinline__ float wred_sum(float v) {
#pragma unroll
  for (int i = 16; i > 0; i >>= 1) v += __shfl_xor(v, i, 32);
  return v;
}
static __device__ __forceinline__ float wred_max(float v) {
#pragma unroll
  for (int i = 16; i > 0; i >>= 1) v = fmaxf(v, __shfl_xor(v, i, 32));
  return v;
}

// ---------------- k0: elementwise fp32 -> bf16 ----------------
__global__ void f32_to_bf16(const float* __restrict__ s, bf16_t* __restrict__ d,
                            int n) {
  int i = blockIdx.x * blockDim.x + threadIdx.x;
  if (i < n) d[i] = (bf16_t)s[i];
}

// ---------------- k1: QKV projection GEMM ----------------
// rows = 4096, cols = 2304, K = 768. Wave tile 32x64 (2 A x 4 B = 8 WMMA/k).
// WG = 8 waves: 2 row-blocks x 4 col-blocks -> 64 x 256 per workgroup.
__global__ __launch_bounds__(256) void qkv_gemm(
    const bf16_t* __restrict__ xb, const bf16_t* __restrict__ wb,
    bf16_t* __restrict__ qo, bf16_t* __restrict__ ko, bf16_t* __restrict__ vto) {
  const int lane = threadIdx.x & 31;
  const int wave = threadIdx.x >> 5;
  const int hl = lane & 15;
  const bool lo = lane < 16;
  const int row0 = blockIdx.x * 64 + (wave & 1) * 32;
  const int col0 = blockIdx.y * 256 + (wave >> 1) * 64;

  const bf16_t* a0p = xb + (size_t)(row0 + hl) * NC + (lo ? 0 : 16);
  const bf16_t* a1p = a0p + (size_t)16 * NC;
  const bf16_t* wp  = wb + (size_t)(col0 + hl) * NC + (lo ? 0 : 16);

  v8f acc[2][4] = {};
  for (int kb = 0; kb < NC; kb += 32) {
    v16bf a0 = ld_split8(a0p + kb);
    v16bf a1 = ld_split8(a1p + kb);
#pragma unroll
    for (int t = 0; t < 4; ++t) {
      v16bf bt = ld_cont16(wp + (size_t)t * 16 * NC + kb);
      acc[0][t] = wmma_bf16(a0, bt, acc[0][t]);
      acc[1][t] = wmma_bf16(a1, bt, acc[1][t]);
    }
  }
#pragma unroll
  for (int mt = 0; mt < 2; ++mt) {
#pragma unroll
    for (int t = 0; t < 4; ++t) {
      v8f a = acc[mt][t];
      const int f = col0 + t * 16 + hl;        // output column of this lane
      const int s = f / NC, rem = f % NC;
      const int h = rem >> 6, d = rem & 63;
#pragma unroll
      for (int i = 0; i < 8; ++i) {
        const int mm = row0 + mt * 16 + (lo ? i : i + 8);
        const int b = mm >> 11, n = mm & (NSEQ - 1);
        bf16_t bv = (bf16_t)a[i];
        if (s == 0)      qo[(((size_t)(b * NH + h) * NSEQ) + n) * HD + d] = bv;
        else if (s == 1) ko[(((size_t)(b * NH + h) * NSEQ) + n) * HD + d] = bv;
        else             vto[(((size_t)(b * NH + h) * HD) + d) * NSEQ + n] = bv;
      }
    }
  }
}

// ---------------- k2: fused scores + double softmax + P@V ----------------
// grid = (N/16, B*H), block = 256 threads (8 wave32s).
// LDS: fp32 scores[16][LDST] + 16 scales + bf16 probs[16][PBS]  (~197 KB)
__global__ __launch_bounds__(256) void spa_attn(
    const bf16_t* __restrict__ q, const bf16_t* __restrict__ k,
    const bf16_t* __restrict__ vt, bf16_t* __restrict__ ao) {
  extern __shared__ float sc[];
  float*  rscale = sc + 16 * LDST;
  bf16_t* pb = (bf16_t*)(rscale + 16);

  const int lane = threadIdx.x & 31;
  const int wave = threadIdx.x >> 5;
  const int hl = lane & 15;
  const bool lo = lane < 16;
  const int bh = blockIdx.y;        // b*NH + h
  const int qb = blockIdx.x * 16;   // first query row of this block

  const bf16_t* qrow = q + ((size_t)bh * NSEQ + qb + hl) * HD + (lo ? 0 : 16);
  const bf16_t* kbp  = k + (size_t)bh * NSEQ * HD;
  const bf16_t* vbp  = vt + (size_t)bh * HD * NSEQ;

  // ---- phase 1: scores S = Q K^T into LDS (128 column tiles over 8 waves)
  {
    v16bf a0 = ld_split8(qrow);        // d = 0..31
    v16bf a1 = ld_split8(qrow + 32);   // d = 32..63
    const int boff = lo ? 0 : 16;
    for (int jt = wave; jt < NSEQ / 16; jt += 8) {
      const bf16_t* krow = kbp + (size_t)(jt * 16 + hl) * HD + boff;
      if (jt + 8 < NSEQ / 16)
        __builtin_prefetch(kbp + (size_t)((jt + 8) * 16 + hl) * HD + boff, 0, 3);
      v16bf b0 = ld_cont16(krow);
      v16bf b1 = ld_cont16(krow + 32);
      v8f c = {};
      c = wmma_bf16(a0, b0, c);
      c = wmma_bf16(a1, b1, c);
      const int col = jt * 16 + hl;
#pragma unroll
      for (int i = 0; i < 8; ++i)
        sc[(lo ? i : i + 8) * LDST + col] = c[i];
    }
  }
  __syncthreads();

  // ---- phase 2: softmax -> periodic mask -> softmax; one wave per row.
  // Unnormalized 2nd-softmax probs are written ONCE as bf16 (A-fragment food);
  // 1/sum2 is folded into the phase-3 epilogue via rscale.
  for (int r = wave; r < 16; r += 8) {
    float*  row  = sc + r * LDST;
    bf16_t* prow = pb + r * PBS;
    const int qmod = (qb + r) & 511;      // N/4 = 512 period
    float m1 = -3.4e38f;
    for (int j = lane; j < NSEQ; j += 32) m1 = fmaxf(m1, row[j] * ATT_SCALE);
    m1 = wred_max(m1);
    float s1 = 0.f, mt = -3.4e38f;
    for (int j = lane; j < NSEQ; j += 32) {
      float e = __expf(row[j] * ATT_SCALE - m1);
      s1 += e;
      row[j] = e;
      float mk = ((j & 511) == qmod) ? 0.5f : 1.0f;
      mt = fmaxf(mt, e * mk);
    }
    s1 = wred_sum(s1);
    mt = wred_max(mt);
    const float inv1 = 1.0f / s1;
    const float m2 = mt * inv1;           // max of p*mask
    float s2 = 0.f;
    for (int j = lane; j < NSEQ; j += 32) {
      float mk = ((j & 511) == qmod) ? 0.5f : 1.0f;
      float u = __expf(row[j] * inv1 * mk - m2);
      s2 += u;
      prow[j] = (bf16_t)u;
    }
    s2 = wred_sum(s2);
    if (lane == 0) rscale[r] = 1.0f / s2;
  }
  __syncthreads();

  // ---- phase 3: out = P @ V ; 4 d-tiles x 2 y-halves over 8 waves,
  //      partial sums combined through the (now dead) fp32 score region.
  {
    const int dt = wave & 3;              // 16-wide d tile
    const int yh = wave >> 2;             // y half: [yh*1024, yh*1024+1024)
    const bf16_t* ap   = pb + (size_t)hl * PBS + (lo ? 0 : 16);
    const bf16_t* vrow = vbp + (size_t)(dt * 16 + hl) * NSEQ + (lo ? 0 : 16);
    v8f c = {};
    const int y0 = yh * (NSEQ / 2);
    for (int yb = y0; yb < y0 + NSEQ / 2; yb += 32) {
      v16bf a = ld_split8(ap + yb);       // ds_load_b128 x2, no cvts
      v16bf b = ld_cont16(vrow + yb);
      c = wmma_bf16(a, b, c);
    }
    __syncthreads();                      // scores are dead; reuse as scratch
    float* scratch = sc;
    if (yh == 1) {
#pragma unroll
      for (int i = 0; i < 8; ++i) scratch[((dt * 32 + lane) << 3) + i] = c[i];
    }
    __syncthreads();
    if (yh == 0) {
#pragma unroll
      for (int i = 0; i < 8; ++i) c[i] += scratch[((dt * 32 + lane) << 3) + i];
      const int b = bh / NH, h = bh % NH;
      const int dcol = dt * 16 + hl;
#pragma unroll
      for (int i = 0; i < 8; ++i) {
        const int mm = lo ? i : i + 8;
        float val = c[i] * rscale[mm];
        ao[((size_t)(b * NSEQ + qb + mm)) * NC + h * HD + dcol] = (bf16_t)val;
      }
    }
  }
}

// ---------------- k3: output projection + bias ----------------
__global__ __launch_bounds__(256) void out_gemm(
    const bf16_t* __restrict__ ab, const bf16_t* __restrict__ wb,
    const float* __restrict__ bias, float* __restrict__ out) {
  const int lane = threadIdx.x & 31;
  const int wave = threadIdx.x >> 5;
  const int hl = lane & 15;
  const bool lo = lane < 16;
  const int row0 = blockIdx.x * 64 + (wave & 1) * 32;
  const int col0 = blockIdx.y * 256 + (wave >> 1) * 64;

  const bf16_t* a0p = ab + (size_t)(row0 + hl) * NC + (lo ? 0 : 16);
  const bf16_t* a1p = a0p + (size_t)16 * NC;
  const bf16_t* wp  = wb + (size_t)(col0 + hl) * NC + (lo ? 0 : 16);

  v8f acc[2][4] = {};
  for (int kb = 0; kb < NC; kb += 32) {
    v16bf a0 = ld_split8(a0p + kb);
    v16bf a1 = ld_split8(a1p + kb);
#pragma unroll
    for (int t = 0; t < 4; ++t) {
      v16bf bt = ld_cont16(wp + (size_t)t * 16 * NC + kb);
      acc[0][t] = wmma_bf16(a0, bt, acc[0][t]);
      acc[1][t] = wmma_bf16(a1, bt, acc[1][t]);
    }
  }
#pragma unroll
  for (int mt = 0; mt < 2; ++mt) {
#pragma unroll
    for (int t = 0; t < 4; ++t) {
      v8f a = acc[mt][t];
      const int o = col0 + t * 16 + hl;
      const float bv = bias[o];
#pragma unroll
      for (int i = 0; i < 8; ++i) {
        const int mm = row0 + mt * 16 + (lo ? i : i + 8);
        out[(size_t)mm * NC + o] = a[i] + bv;
      }
    }
  }
}

// ---------------- host ----------------
extern "C" void kernel_launch(void* const* d_in, const int* in_sizes, int n_in,
                              void* d_out, int out_size, void* d_ws, size_t ws_size,
                              hipStream_t stream) {
  (void)in_sizes; (void)n_in; (void)out_size; (void)ws_size;
  const float* x  = (const float*)d_in[0];
  const float* wq = (const float*)d_in[1];
  const float* wo = (const float*)d_in[2];
  const float* bo = (const float*)d_in[3];
  float* out = (float*)d_out;

  bf16_t* p = (bf16_t*)d_ws;
  bf16_t* xb  = p; p += (size_t)NROWS * NC;          // x in bf16
  bf16_t* wqb = p; p += (size_t)F3C * NC;            // w_qkv bf16
  bf16_t* wob = p; p += (size_t)NC * NC;             // w_out bf16
  bf16_t* qd  = p; p += (size_t)NB * NH * NSEQ * HD; // Q [B,H,N,D]
  bf16_t* kd  = p; p += (size_t)NB * NH * NSEQ * HD; // K [B,H,N,D]
  bf16_t* vtd = p; p += (size_t)NB * NH * NSEQ * HD; // V^T [B,H,D,N]
  bf16_t* aod = p;                                   // attn out [B,N,C] bf16

  const int nx = NROWS * NC, nwq = F3C * NC, nwo = NC * NC;
  f32_to_bf16<<<(nx  + 255) / 256, 256, 0, stream>>>(x,  xb,  nx);
  f32_to_bf16<<<(nwq + 255) / 256, 256, 0, stream>>>(wq, wqb, nwq);
  f32_to_bf16<<<(nwo + 255) / 256, 256, 0, stream>>>(wo, wob, nwo);

  qkv_gemm<<<dim3(NROWS / 64, F3C / 256), 256, 0, stream>>>(xb, wqb, qd, kd, vtd);

  const size_t smem = (size_t)(16 * LDST + 16) * sizeof(float)
                    + (size_t)16 * PBS * sizeof(bf16_t);   // ~197 KB
  (void)hipFuncSetAttribute(reinterpret_cast<const void*>(spa_attn),
                            hipFuncAttributeMaxDynamicSharedMemorySize,
                            (int)smem);
  spa_attn<<<dim3(NSEQ / 16, NB * NH), 256, smem, stream>>>(qd, kd, vtd, aod);

  out_gemm<<<dim3(NROWS / 64, NC / 256), 256, 0, stream>>>(aod, wob, bo, out);
}
